// single_node_actor_17781164605717
// MI455X (gfx1250) — matrix operand order repaired and checked by reading, not compiled, not run on hardware
//
#include <hip/hip_runtime.h>
#include <hip/hip_bf16.h>
#include <math.h>

typedef __attribute__((ext_vector_type(16))) _Float16 v16h;
typedef __attribute__((ext_vector_type(8)))  _Float16 v8h;
typedef __attribute__((ext_vector_type(8)))  float    v8f;

#define SP_BIAS  0.54132485461292f   // log(expm1(1.0))
#define SCALE_LB 1e-4f

// CDNA5 has a hardware tanh (v_tanh_f32, TRANS32 class). Use it when the
// toolchain declares the builtin; fall back to OCML tanhf otherwise.
__device__ __forceinline__ float dev_tanh(float x) {
#if defined(__has_builtin)
#if __has_builtin(__builtin_amdgcn_tanhf)
  return __builtin_amdgcn_tanhf(x);
#elif __has_builtin(__builtin_amdgcn_tanh_f32)
  return __builtin_amdgcn_tanh_f32(x);
#else
  return tanhf(x);
#endif
#else
  return tanhf(x);
#endif
}

// ---------------------------------------------------------------------------
// Kernel 1: h = x @ W1 + b1     ([N,27] @ [27,64] -> [N,64] fp32)
// 4 threads per node, 16 output columns each. W1 rows stay hot in L2.
// ---------------------------------------------------------------------------
__global__ __launch_bounds__(256) void k_linear1(
    const float* __restrict__ x, const float* __restrict__ W1,
    const float* __restrict__ b1, float* __restrict__ h, int N)
{
  int tid = blockIdx.x * 256 + threadIdx.x;
  if (tid >= N * 4) return;
  int node = tid >> 2;
  int q = (tid & 3) * 16;
  const float* xr = x + (size_t)node * 27;
  float acc[16];
#pragma unroll
  for (int j = 0; j < 16; ++j) acc[j] = b1[q + j];
  for (int k = 0; k < 27; ++k) {
    float xv = xr[k];
    const float* wr = W1 + k * 64 + q;
#pragma unroll
    for (int j = 0; j < 16; ++j) acc[j] = fmaf(xv, wr[j], acc[j]);
  }
  float4* dst = (float4*)(h + (size_t)node * 64 + q);
#pragma unroll
  for (int v = 0; v < 4; ++v)
    dst[v] = make_float4(acc[4*v], acc[4*v+1], acc[4*v+2], acc[4*v+3]);
}

// ---------------------------------------------------------------------------
// Kernel 2: agg[dst] += h[src]  (segment_sum over 1.6M edges, 64 channels)
// One thread per (edge, 4-channel quad): float4 gather + 4 HW fp32 atomics.
// Each 16-lane half of a wave covers one edge's 64 channels -> both the gather and the
// atomic target are contiguous 256B regions; indices broadcast from cache.
// This is the roofline-dominant stage (~820 MB of HBM traffic @ 23.3 TB/s).
// ---------------------------------------------------------------------------
__global__ __launch_bounds__(256) void k_scatter(
    const long long* __restrict__ src, const long long* __restrict__ dst,
    const float* __restrict__ h, float* __restrict__ agg, int E)
{
  long long tid = (long long)blockIdx.x * 256 + threadIdx.x;
  if (tid >= (long long)E * 16) return;
  int e = (int)(tid >> 4);
  int q = ((int)tid & 15) * 4;
  long long s = src[e];
  long long d = dst[e];
  float4 v = *(const float4*)(h + s * 64 + q);
  float* a = agg + d * 64 + q;
  __hip_atomic_fetch_add(a + 0, v.x, __ATOMIC_RELAXED, __HIP_MEMORY_SCOPE_AGENT);
  __hip_atomic_fetch_add(a + 1, v.y, __ATOMIC_RELAXED, __HIP_MEMORY_SCOPE_AGENT);
  __hip_atomic_fetch_add(a + 2, v.z, __ATOMIC_RELAXED, __HIP_MEMORY_SCOPE_AGENT);
  __hip_atomic_fetch_add(a + 3, v.w, __ATOMIC_RELAXED, __HIP_MEMORY_SCOPE_AGENT);
}

// ---------------------------------------------------------------------------
// WMMA helpers (CDNA5 16x16x32 f16 -> f32)
// ---------------------------------------------------------------------------
__device__ __forceinline__ v8f wmma16(v16h a, v16h b, v8f c) {
  // (neg_a, A, neg_b, B, c_mod, C, reuse_a, reuse_b)
  return __builtin_amdgcn_wmma_f32_16x16x32_f16(false, a, false, b,
                                                (short)0, c, false, false);
}

// A fragment (16x32 f16) from an fp32 row: per ISA layout,
// lane group 0 holds K = kbase + {0..7, 16..23}, group 1 holds +8 on each run.
__device__ __forceinline__ v16h load_a_frag(const float* __restrict__ rowp,
                                            int kc, int sel) {
  int b0 = kc * 32 + sel * 8;
  float4 x0 = *(const float4*)(rowp + b0);
  float4 x1 = *(const float4*)(rowp + b0 + 4);
  float4 y0 = *(const float4*)(rowp + b0 + 16);
  float4 y1 = *(const float4*)(rowp + b0 + 20);
  v16h a;
  a[0]=(_Float16)x0.x;  a[1]=(_Float16)x0.y;  a[2]=(_Float16)x0.z;  a[3]=(_Float16)x0.w;
  a[4]=(_Float16)x1.x;  a[5]=(_Float16)x1.y;  a[6]=(_Float16)x1.z;  a[7]=(_Float16)x1.w;
  a[8]=(_Float16)y0.x;  a[9]=(_Float16)y0.y;  a[10]=(_Float16)y0.z; a[11]=(_Float16)y0.w;
  a[12]=(_Float16)y1.x; a[13]=(_Float16)y1.y; a[14]=(_Float16)y1.z; a[15]=(_Float16)y1.w;
  return a;
}

// B fragment from pre-swizzled LDS: 32 contiguous bytes per lane.
__device__ __forceinline__ v16h load_b_frag(const _Float16* sB, int f, int lane) {
  const v8h* p = (const v8h*)(sB + f * 512 + lane * 16);
  v8h lo = p[0], hi = p[1];
  v16h b;
#pragma unroll
  for (int i = 0; i < 8; ++i) { b[i] = lo[i]; b[8 + i] = hi[i]; }
  return b;
}

// ---------------------------------------------------------------------------
// Kernel 3: fused  h2 = tanh(agg@Wrel + brel + h@Wroot)
//                  o  = tanh(h2@W2 + b2)
//                  loc = o[:, :8].T ; scale = clip(softplus(o[:,8:]+bias)).T
// One wave32 per 16-node tile; 18 v_wmma per tile.
// ---------------------------------------------------------------------------
__global__ __launch_bounds__(256) void k_graphconv_head(
    const float* __restrict__ h, const float* __restrict__ agg,
    const float* __restrict__ Wrel, const float* __restrict__ brel,
    const float* __restrict__ Wroot, const float* __restrict__ W2,
    const float* __restrict__ b2, float* __restrict__ out, int N)
{
  // 18 B-fragments: [Wrel kc*4+nb : 0..7][Wroot : 8..15][W2 kc : 16..17]
  __shared__ __align__(16) _Float16 sB[18 * 512];
  // per-wave 16x64 f16 scratch tile for the stage1 -> stage2 re-swizzle
  __shared__ __align__(16) _Float16 sT[8][16 * 64];

  // cooperative pre-swizzle of weights into B-fragment layout
  for (int e = threadIdx.x; e < 18 * 512; e += 256) {
    int f = e >> 9;
    int idx = e & 511;
    int bl = idx >> 4;      // lane that owns this element
    int j  = idx & 15;      // half index within the lane's fragment
    float v;
    if (f < 16) {
      const float* Wm = (f & 8) ? Wroot : Wrel;
      int kc = (f >> 2) & 1;
      int nb = f & 3;
      int col = nb * 16 + (bl & 15);
      int K   = kc * 32 + (bl >> 4) * 8 + (j & 7) + (j >> 3) * 16;   // B: group0 K {0..7,16..23}, group1 K {8..15,24..31}
      v = Wm[K * 64 + col];
    } else {
      int kc = f - 16;
      int col = bl & 15;
      int K   = kc * 32 + (bl >> 4) * 8 + (j & 7) + (j >> 3) * 16;
      v = W2[K * 16 + col];
    }
    sB[e] = (_Float16)v;
  }
  __syncthreads();

  const int wave = threadIdx.x >> 5;
  const int lane = threadIdx.x & 31;
  const int lm = lane & 15;
  const int lg = lane >> 4;
  const int nTiles = N >> 4;
  _Float16* tl = sT[wave];

  for (int t = blockIdx.x * 8 + wave; t < nTiles; t += gridDim.x * 8) {
    const size_t rowOff = (size_t)(t * 16 + lm) * 64;
    v16h aAgg[2], aH[2];
#pragma unroll
    for (int kc = 0; kc < 2; ++kc) {
      aAgg[kc] = load_a_frag(agg + rowOff, kc, lg);
      aH[kc]   = load_a_frag(h + rowOff, kc, lg);
    }

    v8f acc[4];
#pragma unroll
    for (int nb = 0; nb < 4; ++nb) {
      float bv = brel[nb * 16 + lm];        // bias per output column
#pragma unroll
      for (int r = 0; r < 8; ++r) acc[nb][r] = bv;
    }

#pragma unroll
    for (int kc = 0; kc < 2; ++kc) {
#pragma unroll
      for (int nb = 0; nb < 4; ++nb) {
        acc[nb] = wmma16(aAgg[kc], load_b_frag(sB, kc * 4 + nb, lane), acc[nb]);
        acc[nb] = wmma16(aH[kc],   load_b_frag(sB, 8 + kc * 4 + nb, lane), acc[nb]);
      }
    }

    // tanh, then stash tile in LDS (row-major f16) for the stage-2 A swizzle.
    // C/D layout: vgpr r, lane lg*16+lm -> M = r + 8*lg, Ncol = lm.
#pragma unroll
    for (int nb = 0; nb < 4; ++nb) {
      int col = nb * 16 + lm;
#pragma unroll
      for (int r = 0; r < 8; ++r)
        tl[(lg * 8 + r) * 64 + col] = (_Float16)dev_tanh(acc[nb][r]);
    }

    // stage 2: [16,64] @ [64,16]
    v8f o;
    {
      float bv = b2[lm];
#pragma unroll
      for (int r = 0; r < 8; ++r) o[r] = bv;
    }
#pragma unroll
    for (int kc = 0; kc < 2; ++kc) {
      int b0 = kc * 32 + lg * 8;
      const v8h* q0 = (const v8h*)(tl + lm * 64 + b0);
      const v8h* q1 = (const v8h*)(tl + lm * 64 + b0 + 16);
      v8h lo = *q0, hi = *q1;
      v16h a;
#pragma unroll
      for (int i = 0; i < 8; ++i) { a[i] = lo[i]; a[8 + i] = hi[i]; }
      o = wmma16(a, load_b_frag(sB, 16 + kc, lane), o);
    }

    // Final: lane owns column lm for 8 consecutive nodes -> contiguous stores.
    // loc cols 0..7 -> out[col*N + node]; scale cols 8..15 -> out[8N + (col-8)*N + node]
    // which is exactly out[col*N + node] as well.
    int nodeBase = t * 16 + lg * 8;
    float vals[8];
    if (lm < 8) {
#pragma unroll
      for (int r = 0; r < 8; ++r) vals[r] = dev_tanh(o[r]);
    } else {
#pragma unroll
      for (int r = 0; r < 8; ++r) {
        float raw = dev_tanh(o[r]) + SP_BIAS;
        vals[r] = fmaxf(log1pf(expf(raw)), SCALE_LB);
      }
    }
    float* dstp = out + (size_t)lm * N + nodeBase;
    ((float4*)dstp)[0] = make_float4(vals[0], vals[1], vals[2], vals[3]);
    ((float4*)dstp)[1] = make_float4(vals[4], vals[5], vals[6], vals[7]);
  }
}

// ---------------------------------------------------------------------------
extern "C" void kernel_launch(void* const* d_in, const int* in_sizes, int n_in,
                              void* d_out, int out_size, void* d_ws, size_t ws_size,
                              hipStream_t stream)
{
  const float*     x     = (const float*)d_in[0];
  const long long* ei    = (const long long*)d_in[1];
  const float*     W1    = (const float*)d_in[2];
  const float*     b1    = (const float*)d_in[3];
  const float*     Wrel  = (const float*)d_in[4];
  const float*     brel  = (const float*)d_in[5];
  const float*     Wroot = (const float*)d_in[6];
  const float*     W2    = (const float*)d_in[7];
  const float*     b2    = (const float*)d_in[8];
  float* out = (float*)d_out;

  const int N = in_sizes[0] / 27;
  const int E = in_sizes[1] / 2;
  const long long* src = ei;       // edge_index[0]
  const long long* dst = ei + E;   // edge_index[1]

  float* h   = (float*)d_ws;                 // [N,64] fp32
  float* agg = h + (size_t)N * 64;           // [N,64] fp32

  hipMemsetAsync(agg, 0, (size_t)N * 64 * sizeof(float), stream);

  {
    int total = N * 4;
    k_linear1<<<(total + 255) / 256, 256, 0, stream>>>(x, W1, b1, h, N);
  }
  {
    long long total = (long long)E * 16;
    int blocks = (int)((total + 255) / 256);
    k_scatter<<<blocks, 256, 0, stream>>>(src, dst, h, agg, E);
  }
  {
    int nTiles = N / 16;
    int blocks = (nTiles + 7) / 8;   // 8 waves (tiles) per 256-thread block
    k_graphconv_head<<<blocks, 256, 0, stream>>>(h, agg, Wrel, brel, Wroot,
                                                 W2, b2, out, N);
  }
}
